// RecommenderModule_1821066134199
// MI455X (gfx1250) — matrix operand order repaired and checked by reading, not compiled, not run on hardware
//
#include <hip/hip_runtime.h>
#include <math.h>

// ---------------------------------------------------------------------------
// Fused CE loss: logits = A @ E^T + bias ; loss = mean(lse(logits) - logits[y])
// A: [2048,128] f32, E: [100000,128] f32, bias: [100000] f32, y: [2048] i32
// bf16 WMMA GEMM fused with flash-style online logsumexp, run in the log2
// domain (log2e folded into the A-fragment conversion) using the raw
// v_exp_f32 / v_log_f32 builtins (no denorm guard).
// Fast path: E pre-converted to bf16 in ws, chunks streamed to double-buffered
// LDS with global_load_async_to_lds_b128 (ASYNCcnt).
// ---------------------------------------------------------------------------

typedef __attribute__((ext_vector_type(16))) __bf16 v16bf;
typedef __attribute__((ext_vector_type(8)))  float  v8f;

#define B_ROWS   2048
#define DIM      128
#define N_ENT    100000
#define NSPLIT   32                                // N-dimension partitions
#define CHUNK    256                               // entities per LDS chunk
#define NCHUNK   ((N_ENT + CHUNK - 1) / CHUNK)     // 391
#define LSTRIDE  136                               // bf16 elems per LDS row
#define BIG_NEG  (-1.0e30f)
#define LOG2E    1.4426950408889634f
#define LN2      0.6931471805599453f

// raw hardware transcendentals: v_exp_f32 (2^x), v_log_f32 (log2 x)
#define EXP2(x)  __builtin_amdgcn_exp2f(x)
#define LOG2(x)  __builtin_amdgcn_logf(x)

// Workspace layout:
//   floats: [0,2048) t | [2048,4096) per-row loss | [4096,+65536) Pm | (+65536) Ps
//   then (16B aligned): Ebf[100000*128] bf16  (fast path only, ~25.6 MB)
#define WS_FIXED_FLOATS (4096 + 2 * NSPLIT * B_ROWS)

// ---------------------------------------------------------------------------
// Kernel 0 (fast path): E f32 -> bf16. 6250 blocks x 256 thr x 8 elems.
// ---------------------------------------------------------------------------
__global__ __launch_bounds__(256) void convert_e_kernel(
    const float* __restrict__ E, __bf16* __restrict__ Ebf)
{
    const size_t i8 = ((size_t)blockIdx.x * 256 + threadIdx.x) * 8;
    float4 a = *(const float4*)(E + i8);
    float4 b = *(const float4*)(E + i8 + 4);
    union { float4 f; __bf16 h[8]; } pk;
    pk.h[0] = (__bf16)a.x; pk.h[1] = (__bf16)a.y;
    pk.h[2] = (__bf16)a.z; pk.h[3] = (__bf16)a.w;
    pk.h[4] = (__bf16)b.x; pk.h[5] = (__bf16)b.y;
    pk.h[6] = (__bf16)b.z; pk.h[7] = (__bf16)b.w;
    *(float4*)(Ebf + i8) = pk.f;
}

// ---------------------------------------------------------------------------
// Kernel 1: exact fp32 target logits  t[b] = dot(A[b], E[y_b]) + bias[y_b]
// ---------------------------------------------------------------------------
__global__ __launch_bounds__(256) void target_kernel(
    const float* __restrict__ A, const float* __restrict__ E,
    const float* __restrict__ bias, const int* __restrict__ y,
    float* __restrict__ t)
{
    const int wave = threadIdx.x >> 5;
    const int lane = threadIdx.x & 31;
    const int row  = blockIdx.x * 8 + wave;
    if (row >= B_ROWS) return;
    const int cls = y[row];
    const float* a = A + (size_t)row * DIM;
    const float* e = E + (size_t)cls * DIM;
    float acc = 0.0f;
    #pragma unroll
    for (int k = lane; k < DIM; k += 32) acc += a[k] * e[k];
    #pragma unroll
    for (int mask = 16; mask > 0; mask >>= 1) acc += __shfl_xor(acc, mask, 32);
    if (lane == 0) t[row] = acc + bias[cls];
}

// ---------------------------------------------------------------------------
// Kernel 2: WMMA GEMM fused with online (max, sumexp) per row, log2 domain.
// Grid: (B/128, NSPLIT). Block: 256 threads = 8 waves, each wave owns 16 rows.
// USEBF=1: bf16 source + async double-buffered LDS. USEBF=0: f32 sync fill.
// ---------------------------------------------------------------------------
template <bool USEBF>
__global__ __launch_bounds__(256) void ce_gemm_lse_kernel(
    const float* __restrict__ A, const float* __restrict__ E,
    const __bf16* __restrict__ Ebf, const float* __restrict__ bias,
    float* __restrict__ Pm, float* __restrict__ Ps)
{
    __shared__ __bf16 Esh[USEBF ? 2 : 1][CHUNK * LSTRIDE];  // 68 KB per buffer

    const int tid  = threadIdx.x;
    const int lane = tid & 31;
    const int wave = tid >> 5;
    const int part = blockIdx.y;
    const int half = lane >> 4;                    // 0: lanes 0-15, 1: lanes 16-31
    const int n16  = lane & 15;

    // ---- A fragments: 16 rows x K=128, bf16, pre-scaled by log2e.
    // ISA A-layout (16-bit, 16x32): each half-fragment = 8 contiguous K values.
    const int kbA  = half * 8;
    const int row0 = blockIdx.x * 128 + wave * 16;
    const float* Arow = A + (size_t)(row0 + n16) * DIM;

    v16bf afrag[4];
    #pragma unroll
    for (int j = 0; j < 4; ++j) {
        union { v16bf v; __bf16 e[16]; } r;
        const float* p0 = Arow + 32 * j + kbA;
        const float* p1 = p0 + 16;
        #pragma unroll
        for (int i = 0; i < 8; ++i) {
            r.e[i]     = (__bf16)(p0[i] * LOG2E);
            r.e[8 + i] = (__bf16)(p1[i] * LOG2E);
        }
        afrag[j] = r.v;
    }

    // Per-lane running softmax state (log2 domain).
    float m[8], s[8];
    #pragma unroll
    for (int r = 0; r < 8; ++r) { m[r] = BIG_NEG; s[r] = 0.0f; }

    // ISA B-layout (16-bit, 32x16): two 8-bf16 runs per lane (K 8h..8h+7, 16+8h..).
    const int kbB = half * 8;

    const unsigned long long ebase = (unsigned long long)(uintptr_t)Ebf;

    // async fill of buffer b with chunk c: lane copies its own 256B row as
    // 16x global_load_async_to_lds_b128 (GVS mode, saddr = Ebf base).
    auto fill_async = [&](int b, int c) {
        const int col = c * CHUNK + tid;
        // LDS byte offset = low 32 bits of the generic address (ISA 10.2).
        unsigned ldsoff = (unsigned)(uintptr_t)&Esh[b][tid * LSTRIDE];
        if (col < N_ENT) {
            unsigned goff = (unsigned)col * (unsigned)(DIM * 2);
            #pragma unroll
            for (int i = 0; i < 16; ++i) {
                asm volatile("global_load_async_to_lds_b128 %0, %1, %2"
                             :: "v"(ldsoff + 16u * i), "v"(goff + 16u * i),
                                "s"(ebase)
                             : "memory");
            }
        } else {
            float4 z = {0.f, 0.f, 0.f, 0.f};
            #pragma unroll
            for (int k = 0; k < DIM; k += 8)
                *(float4*)(&Esh[b][tid * LSTRIDE + k]) = z;
        }
    };

    int cur = 0;
    if constexpr (USEBF) {                          // prologue: prefetch chunk 0
        if (part < NCHUNK) fill_async(0, part);
        asm volatile("s_wait_asynccnt 0" ::: "memory");
        __syncthreads();
    }

    for (int c = part; c < NCHUNK; c += NSPLIT) {
        const int colBase = c * CHUNK;
        int buf;
        if constexpr (USEBF) {
            buf = cur;
            const int cn = c + NSPLIT;              // kick off next chunk's DMA
            if (cn < NCHUNK) fill_async(cur ^ 1, cn);
        } else {
            buf = 0;
            __syncthreads();
            {   // sync fill: stage one f32 entity row per thread, cvt to bf16
                const int col = colBase + tid;
                const bool valid = (col < N_ENT);
                const float* src = E + (size_t)(valid ? col : 0) * DIM;
                #pragma unroll
                for (int k = 0; k < DIM; k += 8) {
                    union { float4 f; __bf16 h[8]; } pk;
                    if (valid) {
                        float4 x0 = *(const float4*)(src + k);
                        float4 x1 = *(const float4*)(src + k + 4);
                        pk.h[0] = (__bf16)x0.x; pk.h[1] = (__bf16)x0.y;
                        pk.h[2] = (__bf16)x0.z; pk.h[3] = (__bf16)x0.w;
                        pk.h[4] = (__bf16)x1.x; pk.h[5] = (__bf16)x1.y;
                        pk.h[6] = (__bf16)x1.z; pk.h[7] = (__bf16)x1.w;
                    } else {
                        #pragma unroll
                        for (int i = 0; i < 8; ++i) pk.h[i] = (__bf16)0.0f;
                    }
                    *(float4*)(&Esh[0][tid * LSTRIDE + k]) = pk.f;
                }
            }
            __syncthreads();
        }

        const bool fullChunk = (colBase + CHUNK <= N_ENT);  // uniform

        // ---- 16 col-tiles of 16 entities, processed in 4 groups of 4
        #pragma unroll
        for (int g = 0; g < 4; ++g) {
            v8f acc[4];
            #pragma unroll
            for (int t = 0; t < 4; ++t) {
                v8f z = {0.f,0.f,0.f,0.f,0.f,0.f,0.f,0.f};
                acc[t] = z;
            }

            #pragma unroll
            for (int j = 0; j < 4; ++j) {
                #pragma unroll
                for (int t = 0; t < 4; ++t) {
                    const int ent = (g * 4 + t) * 16 + n16;
                    const __bf16* bp = &Esh[buf][ent * LSTRIDE + kbB + 32 * j];
                    union { v16bf v; float4 f[2]; } bf;
                    bf.f[0] = *(const float4*)(bp);       // ds_load_b128
                    bf.f[1] = *(const float4*)(bp + 16);  // ds_load_b128
                    acc[t] = __builtin_amdgcn_wmma_f32_16x16x32_bf16(
                        false, afrag[j], false, bf.v, (short)0, acc[t],
                        false, false);
                }
            }

            if (fullChunk) {
                // ---- unmasked update (390 of 391 chunks)
                float bb[4];
                #pragma unroll
                for (int t = 0; t < 4; ++t)
                    bb[t] = bias[colBase + (g * 4 + t) * 16 + n16] * LOG2E;
                #pragma unroll
                for (int r = 0; r < 8; ++r) {
                    float x0 = acc[0][r] + bb[0];
                    float x1 = acc[1][r] + bb[1];
                    float x2 = acc[2][r] + bb[2];
                    float x3 = acc[3][r] + bb[3];
                    float xm = fmaxf(fmaxf(x0, x1), fmaxf(x2, x3));
                    float mn = fmaxf(m[r], xm);
                    float sc = EXP2(m[r] - mn);
                    s[r] = s[r] * sc + EXP2(x0 - mn) + EXP2(x1 - mn)
                                     + EXP2(x2 - mn) + EXP2(x3 - mn);
                    m[r] = mn;
                }
            } else {
                // ---- masked update (global tail chunk only)
                int   colt[4];
                float bb[4];
                bool  oob[4];
                #pragma unroll
                for (int t = 0; t < 4; ++t) {
                    colt[t] = colBase + (g * 4 + t) * 16 + n16;
                    oob[t]  = (colt[t] >= N_ENT);
                    bb[t]   = oob[t] ? 0.0f : bias[colt[t]] * LOG2E;
                }
                #pragma unroll
                for (int r = 0; r < 8; ++r) {
                    float x0 = oob[0] ? BIG_NEG : acc[0][r] + bb[0];
                    float x1 = oob[1] ? BIG_NEG : acc[1][r] + bb[1];
                    float x2 = oob[2] ? BIG_NEG : acc[2][r] + bb[2];
                    float x3 = oob[3] ? BIG_NEG : acc[3][r] + bb[3];
                    float xm = fmaxf(fmaxf(x0, x1), fmaxf(x2, x3));
                    float mn = fmaxf(m[r], xm);
                    float sc = EXP2(m[r] - mn);
                    s[r] = s[r] * sc + EXP2(x0 - mn) + EXP2(x1 - mn)
                                     + EXP2(x2 - mn) + EXP2(x3 - mn);
                    m[r] = mn;
                }
            }
        }

        if constexpr (USEBF) {
            asm volatile("s_wait_asynccnt 0" ::: "memory");
            __syncthreads();
            cur ^= 1;
        }
    }

    // ---- combine (m,s) across the 16 lanes of each half-wave (butterfly).
    // C-layout: VGPR r, lanes 0-15 -> row r ; lanes 16-31 -> row 8+r.
    #pragma unroll
    for (int r = 0; r < 8; ++r) {
        #pragma unroll
        for (int mask = 1; mask < 16; mask <<= 1) {
            float mo = __shfl_xor(m[r], mask, 32);
            float so = __shfl_xor(s[r], mask, 32);
            float M  = fmaxf(m[r], mo);
            s[r] = s[r] * EXP2(m[r] - M) + so * EXP2(mo - M);
            m[r] = M;
        }
        if (n16 == 0) {
            const int rrow = row0 + half * 8 + r;
            Pm[(size_t)part * B_ROWS + rrow] = m[r];
            Ps[(size_t)part * B_ROWS + rrow] = s[r];
        }
    }
}

// ---------------------------------------------------------------------------
// Kernel 3: merge partition partials per row (log2 domain), subtract target.
// ---------------------------------------------------------------------------
__global__ __launch_bounds__(256) void finalize_rows(
    const float* __restrict__ Pm, const float* __restrict__ Ps,
    const float* __restrict__ t, float* __restrict__ L)
{
    const int b = blockIdx.x * blockDim.x + threadIdx.x;
    if (b >= B_ROWS) return;
    float M = BIG_NEG, s = 0.0f;
    for (int p = 0; p < NSPLIT; ++p) {          // fixed order -> deterministic
        float mo = Pm[(size_t)p * B_ROWS + b];
        float so = Ps[(size_t)p * B_ROWS + b];
        float Mn = fmaxf(M, mo);
        s = s * EXP2(M - Mn) + so * EXP2(mo - Mn);
        M = Mn;
    }
    L[b] = LN2 * (M + LOG2(s)) - t[b];
}

// ---------------------------------------------------------------------------
// Kernel 4: deterministic tree mean over 2048 row losses.
// ---------------------------------------------------------------------------
__global__ __launch_bounds__(256) void reduce_mean(
    const float* __restrict__ L, float* __restrict__ out)
{
    __shared__ float sh[256];
    const int tid = threadIdx.x;
    float a = 0.0f;
    #pragma unroll
    for (int i = 0; i < 8; ++i) a += L[tid + 256 * i];   // fixed order
    sh[tid] = a;
    __syncthreads();
    for (int st = 128; st > 0; st >>= 1) {
        if (tid < st) sh[tid] += sh[tid + st];
        __syncthreads();
    }
    if (tid == 0) out[0] = sh[0] / (float)B_ROWS;
}

// ---------------------------------------------------------------------------
extern "C" void kernel_launch(void* const* d_in, const int* in_sizes, int n_in,
                              void* d_out, int out_size, void* d_ws, size_t ws_size,
                              hipStream_t stream)
{
    const float* A    = (const float*)d_in[0];   // [2048,128]
    const float* E    = (const float*)d_in[1];   // [100000,128]
    const float* bias = (const float*)d_in[2];   // [100000]
    const int*   y    = (const int*)  d_in[3];   // [2048]

    float* ws = (float*)d_ws;
    float* t  = ws;                              // 2048
    float* L  = ws + 2048;                       // 2048
    float* Pm = ws + 4096;                       // 32*2048
    float* Ps = Pm + (size_t)NSPLIT * B_ROWS;    // 32*2048

    const size_t fixedBytes = (size_t)WS_FIXED_FLOATS * sizeof(float);  // 540,672
    const size_t ebfBytes   = (size_t)N_ENT * DIM * 2;                  // 25.6 MB
    __bf16* Ebf = (__bf16*)((char*)d_ws + fixedBytes);                  // 16B aligned
    const bool fast = (ws_size >= fixedBytes + ebfBytes);

    target_kernel<<<B_ROWS / 8, 256, 0, stream>>>(A, E, bias, y, t);

    if (fast) {
        convert_e_kernel<<<(N_ENT * DIM) / (256 * 8), 256, 0, stream>>>(E, Ebf);
        ce_gemm_lse_kernel<true><<<dim3(B_ROWS / 128, NSPLIT), 256, 0, stream>>>(
            A, E, Ebf, bias, Pm, Ps);
    } else {
        ce_gemm_lse_kernel<false><<<dim3(B_ROWS / 128, NSPLIT), 256, 0, stream>>>(
            A, E, Ebf, bias, Pm, Ps);
    }

    finalize_rows<<<B_ROWS / 256, 256, 0, stream>>>(Pm, Ps, t, L);
    reduce_mean  <<<1, 256, 0, stream>>>(L, (float*)d_out);
}
